// Attention_1314259993021
// MI455X (gfx1250) — matrix operand order repaired and checked by reading, not compiled, not run on hardware
//
#include <hip/hip_runtime.h>
#include <hip/hip_bf16.h>

// CDNA5 / gfx1250, wave32. WMMA 16x16x32 bf16 (f32 acc) + TDM tensor_load_to_lds.
typedef __attribute__((ext_vector_type(16))) __bf16 v16bf;
typedef __attribute__((ext_vector_type(8)))  float  v8f;
typedef __attribute__((ext_vector_type(4)))  unsigned int u32x4;
typedef __attribute__((ext_vector_type(8)))  unsigned int u32x8;

#define SEQ    2048
#define DMODEL 1024
#define DN     64
#define BATCH  4

// ---------------------------------------------------------------------------
// Tensor Data Mover: DMA a (rows=32 x cols=64) bf16 tile, row-major, from
// global memory into LDS at byte-offset ldsOff. D# per CDNA5 ISA:
//   group0: count=1 | lds_addr | global_addr[56:0] | type=2
//   group1: data_size=2B, tensor_dim0=64, tensor_dim1=SEQ,
//           tile_dim0=64, tile_dim1=32, tensor_dim0_stride=64
// 2D tile -> VADDR2/VADDR3 omitted (NULL). Tracked by TENSORcnt.
// ---------------------------------------------------------------------------
__device__ __forceinline__ void tdm_load_tile_32x64(unsigned ldsOff,
                                                    const __bf16* gptr)
{
    const unsigned long long ga = (unsigned long long)(uintptr_t)gptr;
    u32x4 g0;
    g0[0] = 1u;                                          // count=1, user mode
    g0[1] = ldsOff;                                      // lds_addr (bytes)
    g0[2] = (unsigned)ga;                                // global_addr[31:0]
    g0[3] = (unsigned)((ga >> 32) & 0x01FFFFFFu)         // global_addr[56:32]
          | (2u << 30);                                  // type = 2 ("image")
    u32x8 g1;
    g1[0] = 1u << 16;                                    // data_size = 1 -> 2 bytes
    g1[1] = (unsigned)DN << 16;                          // tensor_dim0[15:0] @ bits 63:48
    g1[2] = (unsigned)SEQ << 16;                         // dim0 hi=0 | tensor_dim1 lo
    g1[3] = (unsigned)DN << 16;                          // dim1 hi=0 | tile_dim0 = 64
    g1[4] = 32u;                                         // tile_dim1 = 32, tile_dim2 = 0
    g1[5] = (unsigned)DN;                                // tensor_dim0_stride = 64
    g1[6] = 0u;                                          // stride hi | dim1_stride lo
    g1[7] = 0u;
    asm volatile("tensor_load_to_lds %0, %1" :: "s"(g0), "s"(g1) : "memory");
}

__device__ __forceinline__ unsigned lds_byte_offset(const void* p)
{
    // generic -> LDS addrspace(3) -> 32-bit LDS byte offset
    return (unsigned)(uintptr_t)(__attribute__((address_space(3))) const void*)p;
}

// ---------------------------------------------------------------------------
// Projection: out[row, 0..63] = (x[row,:] . W^T + bias) * scale, row tile = 16.
// One wave per 16-row tile. W is [64, 1024] row-major (torch Linear layout).
// ---------------------------------------------------------------------------
__global__ __launch_bounds__(32) void proj_kernel(const float* __restrict__ x,
                                                  const float* __restrict__ w,
                                                  const float* __restrict__ bias,
                                                  __bf16* __restrict__ out,
                                                  float scale)
{
    const int lane = threadIdx.x & 31;
    const int hf   = lane >> 4;    // lane-half (0/1)
    const int lm   = lane & 15;

    const int rowBase = blockIdx.x * 16;

    v8f acc[4] = {};

    for (int k0 = 0; k0 < DMODEL; k0 += 32) {
        // A fragment (16x32 bf16): lane row = rowBase+lm,
        // elements e<8 -> K = k0 + hf*8 + e ; e>=8 -> K = k0 + hf*8 + 16 + (e-8)
        const float* xr = x + (size_t)(rowBase + lm) * DMODEL + k0 + hf * 8;
        v16bf a;
        #pragma unroll
        for (int e = 0; e < 8; ++e) {
            a[e]     = (__bf16)xr[e];
            a[e + 8] = (__bf16)xr[e + 16];
        }
        #pragma unroll
        for (int c = 0; c < 4; ++c) {
            // B fragment (32x16 bf16): N = c*16+lm (output col = W row),
            // K = k0 + hf*8 + e (e<8), k0 + hf*8 + 16 + (e-8) (e>=8), matching A.
            const float* wr = w + (size_t)(c * 16 + lm) * DMODEL + k0 + hf * 8;
            v16bf bm;
            #pragma unroll
            for (int e = 0; e < 16; ++e) bm[e] = (__bf16)wr[e < 8 ? e : e + 8];
            acc[c] = __builtin_amdgcn_wmma_f32_16x16x32_bf16(
                false, a, false, bm, (short)0, acc[c], false, false);
        }
    }

    // C layout: VGPR r, lanes 0-15 -> (row r, col lm); lanes 16-31 -> (row r+8, col lm)
    #pragma unroll
    for (int c = 0; c < 4; ++c) {
        #pragma unroll
        for (int r = 0; r < 8; ++r) {
            const int row = rowBase + r + hf * 8;
            const int col = c * 16 + lm;
            const float v = (acc[c][r] + bias[col]) * scale;
            out[(size_t)row * DN + col] = (__bf16)v;
        }
    }
}

// ---------------------------------------------------------------------------
// Flash attention over projected qp/kp/vp (bf16, [B,S,64]).
// One wave handles 16 query rows; loop over keys in tiles of 32.
// qp already carries the 1/sqrt(64) scale. V tile DMA'd to LDS by the TDM.
// ---------------------------------------------------------------------------
__global__ __launch_bounds__(32) void attn_kernel(const __bf16* __restrict__ qp,
                                                  const __bf16* __restrict__ kp,
                                                  const __bf16* __restrict__ vp,
                                                  const float*  __restrict__ mask,
                                                  float* __restrict__ out)
{
    __shared__ __bf16 ldsP[16][34];    // P tile (16 q x 32 k), padded
    __shared__ __bf16 ldsV[32][DN];    // V tile row-major [key][feat], TDM target

    const int lane = threadIdx.x & 31;
    const int hf   = lane >> 4;
    const int lm   = lane & 15;
    const int b     = blockIdx.y;
    const int qbase = blockIdx.x * 16;

    const unsigned ldsVOff = lds_byte_offset(&ldsV[0][0]);

    // Q A-fragments: d = 0..31 (aq0), 32..63 (aq1)
    const __bf16* qr = qp + (size_t)(b * SEQ + qbase + lm) * DN;
    v16bf aq0, aq1;
    #pragma unroll
    for (int e = 0; e < 8; ++e) {
        const int d0 = hf * 8 + e;
        aq0[e]     = qr[d0];
        aq0[e + 8] = qr[d0 + 16];
        aq1[e]     = qr[32 + d0];
        aq1[e + 8] = qr[32 + d0 + 16];
    }

    float mrun[8], lrun[8];
    #pragma unroll
    for (int r = 0; r < 8; ++r) { mrun[r] = -1e30f; lrun[r] = 0.0f; }
    v8f o[4] = {};

    for (int kt = 0; kt < SEQ; kt += 32) {
        // Kick off the V-tile DMA; it overlaps with QK^T + mask + softmax below.
        tdm_load_tile_32x64(ldsVOff, vp + (size_t)(b * SEQ + kt) * DN);

        // Prefetch next iteration's mask tile (16 rows x 128B): one line per lane.
        if (kt + 32 < SEQ) {
            const float* mnext = mask + ((size_t)b * SEQ + qbase + lm) * SEQ
                               + (kt + 32) + hf * 16;
            __builtin_prefetch(mnext, 0, 1);
        }

        // K B-fragments for two 16-key tiles (N = key, K = feature).
        const __bf16* kr0 = kp + (size_t)(b * SEQ + kt + lm) * DN + hf * 8;
        const __bf16* kr1 = kr0 + (size_t)16 * DN;
        v16bf bk0l, bk0h, bk1l, bk1h;
        #pragma unroll
        for (int e = 0; e < 16; ++e) {
            const int ke = e < 8 ? e : e + 8;
            bk0l[e] = kr0[ke];      bk0h[e] = kr0[32 + ke];
            bk1l[e] = kr1[ke];      bk1h[e] = kr1[32 + ke];
        }

        v8f s0 = {}, s1 = {};
        s0 = __builtin_amdgcn_wmma_f32_16x16x32_bf16(false, aq0, false, bk0l, (short)0, s0, false, false);
        s0 = __builtin_amdgcn_wmma_f32_16x16x32_bf16(false, aq1, false, bk0h, (short)0, s0, false, false);
        s1 = __builtin_amdgcn_wmma_f32_16x16x32_bf16(false, aq0, false, bk1l, (short)0, s1, false, false);
        s1 = __builtin_amdgcn_wmma_f32_16x16x32_bf16(false, aq1, false, bk1h, (short)0, s1, false, false);

        // Additive mask penalty (read once from HBM; dominant traffic).
        {
            const float* mp = mask + ((size_t)b * SEQ + qbase + hf * 8) * SEQ + kt;
            #pragma unroll
            for (int r = 0; r < 8; ++r) {
                const float* mr = mp + (size_t)r * SEQ;
                s0[r] += mr[lm]      * (-1e9f);
                s1[r] += mr[16 + lm] * (-1e9f);
            }
        }

        // Row max over the 32 keys of this tile (16-lane-half = one row's cols).
        float tm[8];
        #pragma unroll
        for (int r = 0; r < 8; ++r) tm[r] = fmaxf(s0[r], s1[r]);
        #pragma unroll
        for (int off = 1; off < 16; off <<= 1) {
            #pragma unroll
            for (int r = 0; r < 8; ++r)
                tm[r] = fmaxf(tm[r], __shfl_xor(tm[r], off, 16));
        }

        // Online softmax update.
        float alpha[8], ts[8];
        #pragma unroll
        for (int r = 0; r < 8; ++r) {
            const float mn = fmaxf(mrun[r], tm[r]);
            alpha[r] = __expf(mrun[r] - mn);
            mrun[r]  = mn;
        }
        #pragma unroll
        for (int r = 0; r < 8; ++r) {
            s0[r] = __expf(s0[r] - mrun[r]);
            s1[r] = __expf(s1[r] - mrun[r]);
            ts[r] = s0[r] + s1[r];
        }
        #pragma unroll
        for (int off = 1; off < 16; off <<= 1) {
            #pragma unroll
            for (int r = 0; r < 8; ++r)
                ts[r] += __shfl_xor(ts[r], off, 16);
        }
        #pragma unroll
        for (int r = 0; r < 8; ++r) lrun[r] = lrun[r] * alpha[r] + ts[r];
        #pragma unroll
        for (int c = 0; c < 4; ++c) {
            #pragma unroll
            for (int r = 0; r < 8; ++r) o[c][r] *= alpha[r];
        }

        // C-layout -> A-layout reshape of P through LDS (bf16).
        #pragma unroll
        for (int r = 0; r < 8; ++r) {
            ldsP[r + hf * 8][lm]      = (__bf16)s0[r];
            ldsP[r + hf * 8][16 + lm] = (__bf16)s1[r];
        }
        __syncthreads();

        v16bf ap;
        #pragma unroll
        for (int e = 0; e < 8; ++e) {
            ap[e]     = ldsP[lm][hf * 8 + e];
            ap[e + 8] = ldsP[lm][hf * 8 + 16 + e];
        }

        // V tile must be in LDS now; drain the tensor DMA.
        __builtin_amdgcn_s_wait_tensorcnt(0);

        // O += P (16x32) * V (32x16 per feature chunk); B fragment: lane = feat
        // column, K = key = hf*8 + e (e<8), hf*8 + 16 + (e-8) (e>=8), matching P.
        #pragma unroll
        for (int c = 0; c < 4; ++c) {
            v16bf bv;
            #pragma unroll
            for (int e = 0; e < 16; ++e) bv[e] = ldsV[hf * 8 + (e < 8 ? e : e + 8)][c * 16 + lm];
            o[c] = __builtin_amdgcn_wmma_f32_16x16x32_bf16(
                false, ap, false, bv, (short)0, o[c], false, false);
        }
        __syncthreads();
    }

    // Normalize and store fp32 output.
    #pragma unroll
    for (int r = 0; r < 8; ++r) {
        const float inv = 1.0f / lrun[r];
        const int row = qbase + r + hf * 8;
        float* op = out + ((size_t)b * SEQ + row) * DN;
        #pragma unroll
        for (int c = 0; c < 4; ++c) op[c * 16 + lm] = o[c][r] * inv;
    }
}

// ---------------------------------------------------------------------------
extern "C" void kernel_launch(void* const* d_in, const int* in_sizes, int n_in,
                              void* d_out, int out_size, void* d_ws, size_t ws_size,
                              hipStream_t stream)
{
    const float* q    = (const float*)d_in[0];
    const float* k    = (const float*)d_in[1];
    const float* v    = (const float*)d_in[2];
    const float* mask = (const float*)d_in[3];
    const float* w_q  = (const float*)d_in[4];
    const float* b_q  = (const float*)d_in[5];
    const float* w_k  = (const float*)d_in[6];
    const float* b_k  = (const float*)d_in[7];
    const float* w_v  = (const float*)d_in[8];
    const float* b_v  = (const float*)d_in[9];

    const size_t projElems = (size_t)BATCH * SEQ * DN;   // 512K bf16 = 1 MB each
    __bf16* qp = (__bf16*)d_ws;
    __bf16* kp = qp + projElems;
    __bf16* vp = kp + projElems;

    const int rowTiles = (BATCH * SEQ) / 16;             // 512
    // Fold 1/sqrt(DN) into qp.
    proj_kernel<<<rowTiles, 32, 0, stream>>>(q, w_q, b_q, qp, 0.125f);
    proj_kernel<<<rowTiles, 32, 0, stream>>>(k, w_k, b_k, kp, 1.0f);
    proj_kernel<<<rowTiles, 32, 0, stream>>>(v, w_v, b_v, vp, 1.0f);

    dim3 grid(SEQ / 16, BATCH);                           // 128 x 4 waves
    attn_kernel<<<grid, 32, 0, stream>>>(qp, kp, vp, mask, (float*)d_out);
}
